// SimpleDCRNNModel_51204600103139
// MI455X (gfx1250) — matrix operand order repaired and checked by reading, not compiled, not run on hardware
//
#include <hip/hip_runtime.h>

#define NUM_NODES 19
#define HIDDEN 128
#define INPUT_DIM 128
#define SEQ_LEN 256
#define BATCH 64
#define NUM_CLASSES 4
#define BCHUNK 16
#define GATES3 (3 * HIDDEN)       // 384
#define COMB (NUM_NODES * HIDDEN) // 2432
#define KT_MLP (COMB / 32)        // 76

typedef __attribute__((ext_vector_type(16))) __bf16 v16bf;
typedef __attribute__((ext_vector_type(8)))  __bf16 v8bf;
typedef __attribute__((ext_vector_type(2)))  __bf16 v2bf;
typedef __attribute__((ext_vector_type(8)))  float  v8f;

__device__ __forceinline__ __bf16 f2bf(float f) {
  unsigned u = __builtin_bit_cast(unsigned, f);
  u += 0x7FFFu + ((u >> 16) & 1u);               // round-to-nearest-even
  unsigned short h = (unsigned short)(u >> 16);
  return __builtin_bit_cast(__bf16, h);
}

// packed f32->bf16 pair (v_cvt_pk_bf16_f32 when available)
__device__ __forceinline__ v2bf f2bf2(float a, float b) {
#if __has_builtin(__builtin_amdgcn_cvt_pk_bf16_f32)
  return __builtin_amdgcn_cvt_pk_bf16_f32(a, b);
#else
  v2bf r; r[0] = f2bf(a); r[1] = f2bf(b); return r;
#endif
}

__device__ __forceinline__ v16bf cat16(v8bf a, v8bf b) {
  return __builtin_shufflevector(a, b, 0,1,2,3,4,5,6,7,8,9,10,11,12,13,14,15);
}

__device__ __forceinline__ v8f wmma_bf16(v16bf a, v16bf b, v8f c) {
  // D(16x16,f32) = A(16x32,bf16) * B(32x16,bf16) + C
  return __builtin_amdgcn_wmma_f32_16x16x32_bf16(false, a, false, b, (short)0, c,
                                                 false, false);
}

__device__ __forceinline__ v8f splat8(float v) {
  v8f r;
#pragma unroll
  for (int i = 0; i < 8; ++i) r[i] = v;
  return r;
}

// fast gate activations: v_exp_f32 + v_rcp_f32 (TRANS ops, co-execute with WMMA)
__device__ __forceinline__ float fsigmoid(float x) {
  return __builtin_amdgcn_rcpf(1.0f + __expf(-x));
}
__device__ __forceinline__ float ftanh_fast(float x) {
  // tanh(x) = 1 - 2/(1 + e^{2x}); exp saturation handles |x| large
  return 1.0f - 2.0f * __builtin_amdgcn_rcpf(1.0f + __expf(2.0f * x));
}

// A-fragment (16x32 bf16, ISA layout): lane(ln,hl) elems 0..7 -> K=kt*32+hl*8+0..7,
// elems 8..15 -> K=kt*32+16+hl*8+0..7 : two contiguous 16B LDS loads.
__device__ __forceinline__ v16bf lds_a_frag(const __bf16* row, int kt, int hl) {
  const __bf16* p = row + kt * 32 + hl * 8;
  v8bf s0 = *(const v8bf*)(p);
  v8bf s1 = *(const v8bf*)(p + 16);
  return cat16(s0, s1);
}

// ---------------------------------------------------------------------------
// Pack fp32 weights into bf16 WMMA B-fragments (lane-contiguous 32B blocks).
// B layout (32x16 bf16): lane ln holds col N=ln&15; elems e -> K = (ln>>4)*8 + (e&7) + (e>>3)*16.
// fragment id (W_ih/W_hh): f = ((n*24 + nt)*4 + kt); 512 bf16 each.
// ---------------------------------------------------------------------------
__global__ __launch_bounds__(256) void pack_kernel(
    const float* __restrict__ Wih, const float* __restrict__ Whh,
    const float* __restrict__ W1,
    __bf16* __restrict__ wihp, __bf16* __restrict__ whhp,
    __bf16* __restrict__ w1p) {
  const int stride = gridDim.x * blockDim.x;
  const int gid = blockIdx.x * blockDim.x + threadIdx.x;

  const int nW = NUM_NODES * GATES3 * INPUT_DIM; // 933888
  for (int idx = gid; idx < nW; idx += stride) {
    int f = idx >> 9;           // /512
    int r = idx & 511;
    int l = r >> 4, e = r & 15;
    int n = f / 96;
    int rem = f - n * 96;
    int nt = rem >> 2, kt = rem & 3;
    int col = nt * 16 + (l & 15);            // output gate column (0..383)
    int k   = kt * 32 + ((l >> 4) << 3) + (e & 7) + ((e >> 3) << 4); // contraction index (0..127)
    size_t src = ((size_t)n * GATES3 + col) * INPUT_DIM + k;
    wihp[idx] = f2bf(Wih[src]);
    whhp[idx] = f2bf(Whh[src]);
  }

  const int nW1 = 8 * KT_MLP * 512; // 311296
  for (int idx = gid; idx < nW1; idx += stride) {
    int f = idx >> 9;
    int r = idx & 511;
    int l = r >> 4, e = r & 15;
    int nt = f / KT_MLP, kt = f - nt * KT_MLP;
    int col = nt * 16 + (l & 15);            // hidden unit (0..127)
    int k   = kt * 32 + ((l >> 4) << 3) + (e & 7) + ((e >> 3) << 4); // 0..2431
    w1p[idx] = f2bf(W1[(size_t)k * 128 + col]);
  }
}

// ---------------------------------------------------------------------------
// Fused GRU recurrence. grid = (BATCH/16, NUM_NODES), block = 256 (8 waves).
// Wave w owns hidden cols [16w,16w+16): computes gate tiles nt = {w, w+8, w+16}.
// Weight B-fragments are loop-invariant -> compiler keeps them VGPR-resident.
// ---------------------------------------------------------------------------
__global__ __launch_bounds__(256) void gru_kernel(
    const float* __restrict__ x,          // [B][N][T][D]
    const float* __restrict__ b_ih, const float* __restrict__ b_hh,
    const __bf16* __restrict__ wihp, const __bf16* __restrict__ whhp,
    float* __restrict__ hout)             // [B][N*H]
{
  __shared__ __align__(16) float  h32[BCHUNK][HIDDEN];
  __shared__ __align__(16) __bf16 hbf[BCHUNK][HIDDEN];
  __shared__ __align__(16) __bf16 xbf[BCHUNK][INPUT_DIM];

  const int n   = blockIdx.y;
  const int b0  = blockIdx.x * BCHUNK;
  const int tid = threadIdx.x;
  const int lane = tid & 31;
  const int wave = tid >> 5;
  const int hl   = lane >> 4;
  const int ln   = lane & 15;
  const int j    = wave * 16 + ln;   // hidden column for this lane

  // h0 = 0
  for (int i = tid; i < BCHUNK * HIDDEN; i += 256) {
    ((float*)h32)[i]  = 0.0f;
    ((__bf16*)hbf)[i] = f2bf(0.0f);
  }

  // gate order r,z,n (PyTorch). Fold biases into the accumulators.
  const float bias_r  = b_ih[n * GATES3 + j]       + b_hh[n * GATES3 + j];
  const float bias_z  = b_ih[n * GATES3 + 128 + j] + b_hh[n * GATES3 + 128 + j];
  const float bias_xn = b_ih[n * GATES3 + 256 + j];
  const float bias_hn = b_hh[n * GATES3 + 256 + j];

  // packed B-fragment bases (L2-resident): 32B per lane per fragment
  const __bf16* wih_r = wihp + ((size_t)(n * 24 + wave)      * 4) * 512 + lane * 16;
  const __bf16* wih_z = wihp + ((size_t)(n * 24 + wave + 8)  * 4) * 512 + lane * 16;
  const __bf16* wih_n = wihp + ((size_t)(n * 24 + wave + 16) * 4) * 512 + lane * 16;
  const __bf16* whh_r = whhp + ((size_t)(n * 24 + wave)      * 4) * 512 + lane * 16;
  const __bf16* whh_z = whhp + ((size_t)(n * 24 + wave + 8)  * 4) * 512 + lane * 16;
  const __bf16* whh_n = whhp + ((size_t)(n * 24 + wave + 16) * 4) * 512 + lane * 16;

  // x tile staging assignment: each thread owns 8 contiguous floats of one row
  const int xr = tid >> 4;
  const int xc = (tid & 15) << 3;
  const float* xsrc =
      x + (((size_t)(b0 + xr) * NUM_NODES + n) * SEQ_LEN) * INPUT_DIM + xc;

  for (int t = 0; t < SEQ_LEN; ++t) {
    // stage x_t into LDS as bf16
    const float* xs = xsrc + (size_t)t * INPUT_DIM;
    float4 a0 = *(const float4*)(xs);
    float4 a1 = *(const float4*)(xs + 4);
    v2bf p0 = f2bf2(a0.x, a0.y), p1 = f2bf2(a0.z, a0.w);
    v2bf p2 = f2bf2(a1.x, a1.y), p3 = f2bf2(a1.z, a1.w);
    v8bf xv = { p0[0], p0[1], p1[0], p1[1], p2[0], p2[1], p3[0], p3[1] };
    *(v8bf*)&xbf[xr][xc] = xv;
    if (t + 1 < SEQ_LEN) __builtin_prefetch(xs + INPUT_DIM, 0, 1);
    __syncthreads();   // x staged + previous h update visible

    v8f acc_r  = splat8(bias_r);
    v8f acc_z  = splat8(bias_z);
    v8f acc_xn = splat8(bias_xn);
    v8f acc_hn = splat8(bias_hn);

#pragma unroll
    for (int kt = 0; kt < 4; ++kt) {
      v16bf ax = lds_a_frag(&xbf[ln][0], kt, hl);
      v16bf ah = lds_a_frag(&hbf[ln][0], kt, hl);
      v16bf bxr = *(const v16bf*)(wih_r + kt * 512);
      v16bf bhr = *(const v16bf*)(whh_r + kt * 512);
      acc_r = wmma_bf16(ax, bxr, acc_r);
      acc_r = wmma_bf16(ah, bhr, acc_r);
      v16bf bxz = *(const v16bf*)(wih_z + kt * 512);
      v16bf bhz = *(const v16bf*)(whh_z + kt * 512);
      acc_z = wmma_bf16(ax, bxz, acc_z);
      acc_z = wmma_bf16(ah, bhz, acc_z);
      v16bf bxn = *(const v16bf*)(wih_n + kt * 512);
      v16bf bhn = *(const v16bf*)(whh_n + kt * 512);
      acc_xn = wmma_bf16(ax, bxn, acc_xn);
      acc_hn = wmma_bf16(ah, bhn, acc_hn);
    }
    __syncthreads();   // all waves finished reading hbf

    // gate math; C/D layout: elem e -> row m = e + 8*hl, col = j (disjoint cells)
#pragma unroll
    for (int e = 0; e < 8; ++e) {
      int m = e + hl * 8;
      float hold = h32[m][j];
      float rg = fsigmoid(acc_r[e]);
      float zg = fsigmoid(acc_z[e]);
      float ng = ftanh_fast(fmaf(rg, acc_hn[e], acc_xn[e]));
      float hn = fmaf(zg, hold - ng, ng);   // (1-z)*n + z*h
      h32[m][j] = hn;
      hbf[m][j] = f2bf(hn);
    }
  }

  // write last hidden state -> combined buffer [B][N*H]
#pragma unroll
  for (int e = 0; e < 8; ++e) {
    int m = e + hl * 8;
    hout[(size_t)(b0 + m) * COMB + n * HIDDEN + j] = h32[m][j];
  }
}

// ---------------------------------------------------------------------------
// MLP head: out = relu(combined @ W1 + b1) @ W2 + b2.  One block, 8 waves.
// ---------------------------------------------------------------------------
__global__ __launch_bounds__(256) void mlp_kernel(
    const float* __restrict__ hcomb,  // [64][2432]
    const __bf16* __restrict__ w1p,   // packed [8][76][512]
    const float* __restrict__ b1,
    const float* __restrict__ W2,     // [128][4]
    const float* __restrict__ b2,
    float* __restrict__ out)          // [64][4]
{
  __shared__ __align__(16) float hid[BATCH][128];
  const int tid = threadIdx.x;
  const int lane = tid & 31, wave = tid >> 5;
  const int hl = lane >> 4, ln = lane & 15;

  v8f acc[4];
  const float bias = b1[wave * 16 + ln];
#pragma unroll
  for (int mt = 0; mt < 4; ++mt) acc[mt] = splat8(bias);

  for (int kt = 0; kt < KT_MLP; ++kt) {
    v16bf bf = *(const v16bf*)(w1p + ((size_t)wave * KT_MLP + kt) * 512 + lane * 16);
#pragma unroll
    for (int mt = 0; mt < 4; ++mt) {
      const float* row = hcomb + (size_t)(mt * 16 + ln) * COMB + kt * 32 + hl * 8;
      float4 a0 = *(const float4*)(row);
      float4 a1 = *(const float4*)(row + 4);
      float4 a2 = *(const float4*)(row + 16);
      float4 a3 = *(const float4*)(row + 20);
      v2bf q0 = f2bf2(a0.x, a0.y), q1 = f2bf2(a0.z, a0.w);
      v2bf q2 = f2bf2(a1.x, a1.y), q3 = f2bf2(a1.z, a1.w);
      v2bf q4 = f2bf2(a2.x, a2.y), q5 = f2bf2(a2.z, a2.w);
      v2bf q6 = f2bf2(a3.x, a3.y), q7 = f2bf2(a3.z, a3.w);
      v16bf af = { q0[0], q0[1], q1[0], q1[1], q2[0], q2[1], q3[0], q3[1],
                   q4[0], q4[1], q5[0], q5[1], q6[0], q6[1], q7[0], q7[1] };
      acc[mt] = wmma_bf16(af, bf, acc[mt]);
    }
  }

#pragma unroll
  for (int mt = 0; mt < 4; ++mt)
#pragma unroll
    for (int e = 0; e < 8; ++e)
      hid[mt * 16 + e + hl * 8][wave * 16 + ln] = fmaxf(acc[mt][e], 0.0f);
  __syncthreads();

  const int b = tid >> 2, cls = tid & 3;   // 256 threads == 64*4 outputs
  float s = b2[cls];
#pragma unroll 8
  for (int k = 0; k < 128; ++k)
    s = fmaf(hid[b][k], W2[k * NUM_CLASSES + cls], s);
  out[b * NUM_CLASSES + cls] = s;
}

// ---------------------------------------------------------------------------
extern "C" void kernel_launch(void* const* d_in, const int* in_sizes, int n_in,
                              void* d_out, int out_size, void* d_ws, size_t ws_size,
                              hipStream_t stream) {
  const float* inputs = (const float*)d_in[0];
  // d_in[1] = seq_lengths (unused, matching reference)
  const float* W_ih = (const float*)d_in[2];
  const float* W_hh = (const float*)d_in[3];
  const float* b_ih = (const float*)d_in[4];
  const float* b_hh = (const float*)d_in[5];
  const float* W1   = (const float*)d_in[6];
  const float* b1   = (const float*)d_in[7];
  const float* W2   = (const float*)d_in[8];
  const float* b2   = (const float*)d_in[9];

  // workspace carve-out (all offsets 256B-aligned):
  //   wihp: 933888 bf16 | whhp: 933888 bf16 | w1p: 311296 bf16 | hout: 155648 f32
  __bf16* wihp = (__bf16*)d_ws;
  __bf16* whhp = wihp + (size_t)NUM_NODES * GATES3 * INPUT_DIM;
  __bf16* w1p  = whhp + (size_t)NUM_NODES * GATES3 * INPUT_DIM;
  float*  hout = (float*)(w1p + (size_t)8 * KT_MLP * 512);

  pack_kernel<<<2048, 256, 0, stream>>>(W_ih, W_hh, W1, wihp, whhp, w1p);
  gru_kernel<<<dim3(BATCH / BCHUNK, NUM_NODES), 256, 0, stream>>>(
      inputs, b_ih, b_hh, wihp, whhp, hout);
  mlp_kernel<<<1, 256, 0, stream>>>(hout, b1 ? w1p : w1p, b1, W2, b2, (float*)d_out);
}
